// MoEBlock_8022998909621
// MI455X (gfx1250) — compile-verified, matrix operand order checked by reading, not run on hardware
//
#include <hip/hip_runtime.h>

// MI455X / gfx1250 fused MoE block.
//  - bf16 WMMA (v_wmma_f32_16x16x32_bf16) for both expert GEMMs, f32 accum
//  - top-2 gating + per-expert token compaction in LDS (avoids 4x dense compute)
//  - each work item = 2 token-subtiles of one expert: every weight A-fragment
//    is loaded once and used by 2 WMMAs -> halves L2 weight traffic (the
//    dominant cost after compaction; HBM floor ~5.8us, WMMA ~8us)
//  - channel-sliced coalesced global I/O, token-major bf16 x tile in LDS
//  - residual + softmax-combine via ds_add_f32 scatter into f32 LDS accumulator

typedef __attribute__((ext_vector_type(16))) __bf16 v16bf;
typedef __attribute__((ext_vector_type(8)))  __bf16 v8bf;
typedef __attribute__((ext_vector_type(8)))  float  v8f;

#define CH    128      // channels
#define TILE  128      // tokens per workgroup
#define NEXP  8        // experts
#define XST   136      // xs row stride (bf16 elems), 16B aligned, bank-skewed
#define AST   132      // acc row stride (f32 elems), bank-skewed
#define SPAT  65536    // t*h*w (per-batch spatial size); channel-plane stride
#define GSUB  2        // token-subtiles per work item (A-fragment reuse factor)

// Weight f32 -> bf16 conversion into workspace (run once per launch; L2-hot after).
__global__ void moe_prep(const float* __restrict__ W1, const float* __restrict__ W2,
                         __bf16* __restrict__ W1b, __bf16* __restrict__ W2b) {
  int i = blockIdx.x * 256 + threadIdx.x;
  if (i < NEXP * CH * CH) {
    W1b[i] = (__bf16)W1[i];
    W2b[i] = (__bf16)W2[i];
  }
}

// Load one 16-lane-striped WMMA fragment half-pair: per lane, 8 contiguous bf16
// at p and 8 at p+16 (matches 16-bit A/B VGPR layout for 16x16x32).
__device__ __forceinline__ v16bf load_frag(const __bf16* p) {
  v8bf lo = *(const v8bf*)(p);
  v8bf hi = *(const v8bf*)(p + 16);
  v16bf r;
#pragma unroll
  for (int i = 0; i < 8; ++i) { r[i] = lo[i]; r[i + 8] = hi[i]; }
  return r;
}

__global__ void __launch_bounds__(256)
moe_main(const float* __restrict__ x,
         const float* __restrict__ gate_w, const float* __restrict__ gate_b,
         const __bf16* __restrict__ W1b, const float* __restrict__ b1,
         const float* __restrict__ alpha,
         const __bf16* __restrict__ W2b, const float* __restrict__ b2,
         float* __restrict__ out)
{
  __shared__ __align__(16) __bf16 xs[TILE * XST];       // token-major x tile (bf16)
  __shared__ float          accs[TILE * AST];           // f32 out accumulator (init = x)
  __shared__ float          gws[CH * NEXP];             // gate weights
  __shared__ float          gbs[NEXP];
  __shared__ int            cnt[NEXP];                  // tokens per expert
  __shared__ unsigned short toks[NEXP][TILE];           // compacted token index lists
  __shared__ float          tprob[NEXP][TILE];          // matching combine probs

  const int tid  = threadIdx.x;
  const long n0  = (long)blockIdx.x * TILE;             // SPAT % TILE == 0 -> no b straddle
  const int bidx = (int)(n0 >> 16);
  const int s0   = (int)(n0 & (SPAT - 1));
  const float* xb = x + (size_t)bidx * CH * SPAT + s0;

  for (int i = tid; i < CH * NEXP; i += 256) gws[i] = gate_w[i];
  if (tid < NEXP) { gbs[tid] = gate_b[tid]; cnt[tid] = 0; }

  // ---- Stage x tile: coalesced over tokens (fixed channel), transpose into LDS ----
#pragma unroll 4
  for (int i = tid; i < CH * TILE; i += 256) {
    int c = i >> 7, n = i & (TILE - 1);
    float v = xb[(size_t)c * SPAT + n];
    accs[n * AST + c] = v;                  // residual init (f32)
    xs[n * XST + c]   = (__bf16)v;          // GEMM operand (bf16)
  }
  __syncthreads();

  // ---- Gating: one thread per token, f32, top-2 + softmax, compact per expert ----
  if (tid < TILE) {
    const int n = tid;
    float lg[NEXP];
#pragma unroll
    for (int e = 0; e < NEXP; ++e) lg[e] = gbs[e];
    for (int c = 0; c < CH; ++c) {
      float xv = accs[n * AST + c];
#pragma unroll
      for (int e = 0; e < NEXP; ++e) lg[e] += xv * gws[c * NEXP + e];
    }
    int e0 = 0; float v0 = lg[0];
#pragma unroll
    for (int e = 1; e < NEXP; ++e) if (lg[e] > v0) { v0 = lg[e]; e0 = e; }
    int e1 = -1; float v1 = -3.0e38f;
#pragma unroll
    for (int e = 0; e < NEXP; ++e) if (e != e0 && lg[e] > v1) { v1 = lg[e]; e1 = e; }
    float t  = __expf(v1 - v0);             // softmax over selected logits
    float p1 = t / (1.0f + t);
    float p0 = 1.0f - p1;
    int q0 = atomicAdd(&cnt[e0], 1); toks[e0][q0] = (unsigned short)n; tprob[e0][q0] = p0;
    int q1 = atomicAdd(&cnt[e1], 1); toks[e1][q1] = (unsigned short)n; tprob[e1][q1] = p1;
  }
  __syncthreads();

  // ---- Expert MLP: one wave per (expert, pair-of-16-token-subtiles) item ----
  const int wave = tid >> 5, lane = tid & 31;
  const int g = lane >> 4, j = lane & 15;   // lane -> (half, column) per WMMA layout
  int nt[NEXP], total = 0;
#pragma unroll
  for (int e = 0; e < NEXP; ++e) { nt[e] = (cnt[e] + (16 * GSUB - 1)) / (16 * GSUB); total += nt[e]; }

  for (int item = wave; item < total; item += 8) {
    int e = 0, s = item;
    while (s >= nt[e]) { s -= nt[e]; ++e; }

    bool  valid[GSUB];
    int   tok[GSUB];
    float p[GSUB];
#pragma unroll
    for (int t = 0; t < GSUB; ++t) {
      const int slot = (s * GSUB + t) * 16 + j;
      valid[t] = slot < cnt[e];
      tok[t]   = valid[t] ? (int)toks[e][slot] : 0;
      p[t]     = valid[t] ? tprob[e][slot] : 0.0f;
    }
    const float alph = alpha[e];

    // B fragments for layer 1: rows = compacted tokens (gather free via index list)
    v16bf B1[GSUB][4];
#pragma unroll
    for (int t = 0; t < GSUB; ++t) {
      const __bf16* xr = xs + tok[t] * XST + 8 * g;
#pragma unroll
      for (int kk = 0; kk < 4; ++kk) B1[t][kk] = load_frag(xr + 32 * kk);
    }

    // Layer 1: h1 = PReLU(W1[e] @ x + b1[e]); each A frag feeds GSUB WMMAs.
    v16bf H[GSUB][4];
    const __bf16* w1e = W1b + e * (CH * CH) + j * CH + 8 * g;
    const float*  b1e = b1 + e * CH;
#pragma unroll
    for (int u = 0; u < 8; ++u) {
      v8f d[GSUB];
#pragma unroll
      for (int t = 0; t < GSUB; ++t) d[t] = (v8f){0.f,0.f,0.f,0.f,0.f,0.f,0.f,0.f};
#pragma unroll
      for (int kk = 0; kk < 4; ++kk) {
        v16bf a = load_frag(w1e + (16 * u) * CH + 32 * kk);
#pragma unroll
        for (int t = 0; t < GSUB; ++t)
          d[t] = __builtin_amdgcn_wmma_f32_16x16x32_bf16(false, a, false, B1[t][kk],
                                                         (short)0, d[t], false, false);
      }
      const float* bb = b1e + 16 * u + 8 * g;
#pragma unroll
      for (int t = 0; t < GSUB; ++t) {
#pragma unroll
        for (int v = 0; v < 8; ++v) {
          float h = d[t][v] + bb[v];
          h = (h > 0.f) ? h : alph * h;                 // scalar PReLU per expert
          H[t][u >> 1][(u & 1) * 8 + v] = (__bf16)h;    // D-layout -> B-frag, lane-local
        }
      }
    }

    // Layer 2: out_e = W2[e] @ h1 + b2[e]; scale by gate prob, scatter-add to acc.
    const __bf16* w2e = W2b + e * (CH * CH) + j * CH + 8 * g;
    const float*  b2e = b2 + e * CH;
#pragma unroll
    for (int u = 0; u < 8; ++u) {
      v8f d[GSUB];
#pragma unroll
      for (int t = 0; t < GSUB; ++t) d[t] = (v8f){0.f,0.f,0.f,0.f,0.f,0.f,0.f,0.f};
#pragma unroll
      for (int kk = 0; kk < 4; ++kk) {
        v16bf a = load_frag(w2e + (16 * u) * CH + 32 * kk);
#pragma unroll
        for (int t = 0; t < GSUB; ++t)
          d[t] = __builtin_amdgcn_wmma_f32_16x16x32_bf16(false, a, false, H[t][kk],
                                                         (short)0, d[t], false, false);
      }
      const float* bb = b2e + 16 * u + 8 * g;
#pragma unroll
      for (int t = 0; t < GSUB; ++t) {
        if (valid[t]) {
          float* ar = accs + tok[t] * AST + 16 * u + 8 * g;
#pragma unroll
          for (int v = 0; v < 8; ++v)
            atomicAdd(&ar[v], p[t] * (d[t][v] + bb[v]));   // ds_add_f32
        }
      }
    }
  }
  __syncthreads();

  // ---- Store: transpose back, coalesced over tokens per channel ----
#pragma unroll 4
  for (int i = tid; i < CH * TILE; i += 256) {
    int c = i >> 7, n = i & (TILE - 1);
    out[((size_t)(bidx * CH + c)) * SPAT + s0 + n] = accs[n * AST + c];
  }
}

extern "C" void kernel_launch(void* const* d_in, const int* in_sizes, int n_in,
                              void* d_out, int out_size, void* d_ws, size_t ws_size,
                              hipStream_t stream) {
  const float* x      = (const float*)d_in[0];   // [2,128,16,64,64]
  const float* gate_w = (const float*)d_in[1];   // [128,8]
  const float* gate_b = (const float*)d_in[2];   // [8]
  const float* W1     = (const float*)d_in[3];   // [8,128,128]
  const float* b1     = (const float*)d_in[4];   // [8,128]
  const float* alpha  = (const float*)d_in[5];   // [8]
  const float* W2     = (const float*)d_in[6];   // [8,128,128]
  const float* b2     = (const float*)d_in[7];   // [8,128]
  // d_in[8] = top_k (== 2, baked into the kernel)
  float* out = (float*)d_out;

  // Workspace: bf16 copies of W1/W2 (512 KB).
  __bf16* W1b = (__bf16*)d_ws;
  __bf16* W2b = W1b + NEXP * CH * CH;

  moe_prep<<<(NEXP * CH * CH + 255) / 256, 256, 0, stream>>>(W1, W2, W1b, W2b);

  const int ntiles = (2 * 16 * 64 * 64) / TILE;  // 131072 tokens / 128 = 1024 WGs
  moe_main<<<ntiles, 256, 0, stream>>>(x, gate_w, gate_b, W1b, b1, alpha, W2b, b2, out);
}
